// CausalAttention_40845138985312
// MI455X (gfx1250) — matrix operand order repaired and checked by reading, not compiled, not run on hardware
//
#include <hip/hip_runtime.h>
#include <hip/hip_bf16.h>

#define B_ 2
#define T_ 2048
#define D_ 2048
#define H_ 16
#define LAT_ 512
#define RD_ 64
#define HD_ 128
#define QKD_ 192   // HD_ + RD_

typedef __bf16 bf16;
typedef __attribute__((ext_vector_type(16))) __bf16 v16bf;
typedef __attribute__((ext_vector_type(8)))  __bf16 v8bf;
typedef __attribute__((ext_vector_type(8)))  float  v8f;

__device__ inline v8f v8f_zero() {
    v8f z = {0.f, 0.f, 0.f, 0.f, 0.f, 0.f, 0.f, 0.f};
    return z;
}

// ---------------------------------------------------------------------------
// Global -> LDS tile copy, 16 bytes per call per lane.
// CDNA5 async-to-LDS path (ASYNCcnt-tracked, no VGPR staging).  The builtin's
// signature (leaked by the round-2 diagnostic) takes int32x4 pointers in
// addrspace(1) (global) and addrspace(3) (LDS), plus imm offset and cpol.
// ---------------------------------------------------------------------------
typedef int v4i_ __attribute__((vector_size(16)));

#if defined(__HIP_DEVICE_COMPILE__) && __has_builtin(__builtin_amdgcn_global_load_async_to_lds_b128)
#define ASYNC_LDS 1
typedef __attribute__((address_space(1))) v4i_ as1_v4i;
typedef __attribute__((address_space(3))) v4i_ as3_v4i;
__device__ inline void lds_copy16(bf16* dst, const bf16* src) {
    __builtin_amdgcn_global_load_async_to_lds_b128(
        (as1_v4i*)const_cast<bf16*>(src),
        (as3_v4i*)dst, 0, 0);
}
__device__ inline void lds_copy_fence() {
#if __has_builtin(__builtin_amdgcn_s_wait_asynccnt)
    __builtin_amdgcn_s_wait_asynccnt(0);
#else
    asm volatile("s_wait_asynccnt 0" ::: "memory");
#endif
}
#else
#define ASYNC_LDS 0
__device__ inline void lds_copy16(bf16* dst, const bf16* src) {
    *reinterpret_cast<v8bf*>(dst) = *reinterpret_cast<const v8bf*>(src);
}
__device__ inline void lds_copy_fence() {}
#endif

// Build an A-fragment (16x32 bf16, MxK) from a row-major row pointer.
// Lanes 0-15 -> M=lane, K elems {0..15}; lanes 16-31 -> K {16..31}.
__device__ inline v16bf load_a_frag(const bf16* row, int sel) {
    int base = sel * 16;
    v8bf lo = *reinterpret_cast<const v8bf*>(row + base);
    v8bf hi = *reinterpret_cast<const v8bf*>(row + base + 8);
    v16bf f;
#pragma unroll
    for (int e = 0; e < 8; ++e) { f[e] = lo[e]; f[8 + e] = hi[e]; }
    return f;
}

// ---------------------------------------------------------------------------
// Elementwise converts
// ---------------------------------------------------------------------------
__global__ void f32_to_bf16_kernel(const float* __restrict__ in,
                                   bf16* __restrict__ out, size_t n) {
    size_t i = (size_t)blockIdx.x * blockDim.x + threadIdx.x;
    if (i < n) out[i] = (bf16)in[i];
}

// in: K x N (row major f32) -> out: N x K (row major bf16)
__global__ void transpose_f32_to_bf16_kernel(const float* __restrict__ in,
                                             bf16* __restrict__ out,
                                             int K, int N) {
    size_t i = (size_t)blockIdx.x * blockDim.x + threadIdx.x;
    if (i < (size_t)K * N) {
        int k = (int)(i / N);
        int n = (int)(i % N);
        out[(size_t)n * K + k] = (bf16)in[i];
    }
}

// ---------------------------------------------------------------------------
// Generic bf16 WMMA GEMM: C[M,N] = A[M,K] (row major) * Bt[N,K] (row major,
// i.e. B transposed).  Block tile 128x64, 8 waves of 32x32, K-step 32.
// All problem dims here are multiples of the tile sizes.
// ---------------------------------------------------------------------------
template <typename OutT>
__global__ void __launch_bounds__(256)
gemm_bf16_kernel(const bf16* __restrict__ A, const bf16* __restrict__ Bt,
                 OutT* __restrict__ C, int M, int N, int K) {
    __shared__ bf16 As[128][32];
    __shared__ bf16 Bs[64][32];
    const int tid  = threadIdx.x;
    const int lane = tid & 31;
    const int wave = tid >> 5;
    const int wm = wave >> 1;      // 0..3
    const int wn = wave & 1;       // 0..1
    const int m0 = blockIdx.y * 128;
    const int n0 = blockIdx.x * 64;
    const int sel = lane >> 4;
    const int ln  = lane & 15;

    v8f acc[2][2];
#pragma unroll
    for (int i = 0; i < 2; ++i)
#pragma unroll
        for (int j = 0; j < 2; ++j) acc[i][j] = v8f_zero();

    for (int k0 = 0; k0 < K; k0 += 32) {
        {   // A tile: 128x32, 2 threads per row, 16 elems each
            int r = tid >> 1, c = (tid & 1) * 16;
            const bf16* src = A + (size_t)(m0 + r) * K + k0 + c;
            if (k0 + 32 < K) __builtin_prefetch(src + 32, 0, 0);
            lds_copy16(&As[r][c],     src);
            lds_copy16(&As[r][c + 8], src + 8);
        }
        {   // Bt tile: 64x32, 4 threads per row, 8 elems each
            int r = tid >> 2, c = (tid & 3) * 8;
            const bf16* src = Bt + (size_t)(n0 + r) * K + k0 + c;
            if (k0 + 32 < K) __builtin_prefetch(src + 32, 0, 0);
            lds_copy16(&Bs[r][c], src);
        }
        lds_copy_fence();
        __syncthreads();

        v16bf af[2], bfr[2];
#pragma unroll
        for (int i = 0; i < 2; ++i)
            af[i] = load_a_frag(&As[wm * 32 + i * 16 + ln][0], sel);
#pragma unroll
        for (int j = 0; j < 2; ++j)
            bfr[j] = *reinterpret_cast<const v16bf*>(&Bs[wn * 32 + j * 16 + ln][sel * 16]);

#pragma unroll
        for (int i = 0; i < 2; ++i)
#pragma unroll
            for (int j = 0; j < 2; ++j)
                acc[i][j] = __builtin_amdgcn_wmma_f32_16x16x32_bf16(
                    false, af[i], false, bfr[j], (short)0, acc[i][j], false, false);
        __syncthreads();
    }

    // C layout: lane = (sel,n); VGPR r -> row r + 8*sel
#pragma unroll
    for (int i = 0; i < 2; ++i)
#pragma unroll
        for (int j = 0; j < 2; ++j)
#pragma unroll
            for (int r = 0; r < 8; ++r) {
                int row = m0 + wm * 32 + i * 16 + r + 8 * sel;
                int col = n0 + wn * 32 + j * 16 + ln;
                C[(size_t)row * N + col] = (OutT)acc[i][j][r];
            }
}

// ---------------------------------------------------------------------------
// Pack Q' = [q | rope(qr)] (B,H,T,192), K' = [k | rope(kr)] (B,H,T,192),
// Vt (B,H,HD,T).  kr is shared across heads.
// ---------------------------------------------------------------------------
__global__ void __launch_bounds__(256)
pack_rope_kernel(const bf16* __restrict__ qtmp, const bf16* __restrict__ qrtmp,
                 const bf16* __restrict__ ktmp, const bf16* __restrict__ vtmp,
                 const bf16* __restrict__ krtmp,
                 bf16* __restrict__ Qp, bf16* __restrict__ Kp,
                 bf16* __restrict__ Vt) {
    const int t = blockIdx.x;
    const int b = blockIdx.y;
    const int tid = threadIdx.x;

    __shared__ float c_tab[RD_], s_tab[RD_], skr[RD_];

    if (tid < RD_) {
        int j = tid & (RD_ / 2 - 1);   // freqs tiled twice
        float freq = __expf(-((float)(2 * j) / (float)RD_) * __logf(10000.0f));
        float sv, cv;
        __sincosf((float)t * freq, &sv, &cv);
        c_tab[tid] = cv;
        s_tab[tid] = sv;
    }
    __syncthreads();
    if (tid < RD_) {
        const bf16* kr = krtmp + (size_t)(b * T_ + t) * RD_;
        float x  = (float)kr[tid];
        float xr = (tid < RD_ / 2) ? -(float)kr[tid + RD_ / 2]
                                   :  (float)kr[tid - RD_ / 2];
        skr[tid] = x * c_tab[tid] + xr * s_tab[tid];
    }
    __syncthreads();

    const size_t rowq = (size_t)(b * T_ + t);
    for (int h = 0; h < H_; ++h) {
        size_t obase = ((size_t)(b * H_ + h) * T_ + t) * QKD_;
        if (tid < HD_) {
            Qp[obase + tid] = qtmp[rowq * D_ + h * HD_ + tid];
            Kp[obase + tid] = ktmp[rowq * D_ + h * HD_ + tid];
            Vt[((size_t)(b * H_ + h) * HD_ + tid) * T_ + t] =
                vtmp[rowq * D_ + h * HD_ + tid];
        } else if (tid < HD_ + RD_) {
            int i = tid - HD_;
            const bf16* qr = qrtmp + rowq * (H_ * RD_) + h * RD_;
            float x  = (float)qr[i];
            float xr = (i < RD_ / 2) ? -(float)qr[i + RD_ / 2]
                                     :  (float)qr[i - RD_ / 2];
            Qp[obase + HD_ + i] = (bf16)(x * c_tab[i] + xr * s_tab[i]);
            Kp[obase + HD_ + i] = (bf16)skr[i];
        }
    }
}

// ---------------------------------------------------------------------------
// Flash attention: grid (T/64, H, B); 4 waves, 16 Q rows per wave,
// 32-key tiles, K-dim 192 through Q'/K', online softmax, O += P*V via WMMA.
// Output written as (B,T,H*HD) bf16 for the final Wo GEMM.
// ---------------------------------------------------------------------------
__global__ void __launch_bounds__(128)
flash_attn_kernel(const bf16* __restrict__ Qp, const bf16* __restrict__ Kp,
                  const bf16* __restrict__ Vt, bf16* __restrict__ Ao) {
    __shared__ bf16 Ks[32][QKD_];     // 32 keys x 192
    __shared__ bf16 Vs[HD_][32];      // [hd][key]  (== Bt layout)
    __shared__ bf16 Ps[4][16][32];    // per-wave P transpose staging

    const int tid  = threadIdx.x;
    const int lane = tid & 31;
    const int wave = tid >> 5;
    const int sel  = lane >> 4;
    const int ln   = lane & 15;
    const int q0   = blockIdx.x * 64;
    const int h    = blockIdx.y;
    const int b    = blockIdx.z;

    const size_t qkbase = (size_t)(b * H_ + h) * T_ * QKD_;
    const size_t vbase  = (size_t)(b * H_ + h) * HD_ * T_;

    // Preload Q fragments: 16 rows x 192 => 6 fragments of K=32
    v16bf qf[6];
    {
        const bf16* qrow = Qp + qkbase + (size_t)(q0 + wave * 16 + ln) * QKD_;
#pragma unroll
        for (int kc = 0; kc < 6; ++kc)
            qf[kc] = load_a_frag(qrow + kc * 32, sel);
    }

    v8f o[8];
#pragma unroll
    for (int i = 0; i < 8; ++i) o[i] = v8f_zero();
    float mrow[8], lrow[8];
#pragma unroll
    for (int r = 0; r < 8; ++r) { mrow[r] = -INFINITY; lrow[r] = 0.0f; }
    const float scale = 0.07216878364870323f;   // 1/sqrt(192)

    const int nkb = (q0 + 64) / 32;
    for (int kb = 0; kb < nkb; ++kb) {
        const int kbase = kb * 32;
        {   // K' tile: 32 rows x 192; 4 threads/row, 48 elems each
            int r = tid >> 2, c = (tid & 3) * 48;
            const bf16* src = Kp + qkbase + (size_t)(kbase + r) * QKD_ + c;
#pragma unroll
            for (int u = 0; u < 6; ++u)
                lds_copy16(&Ks[r][c + u * 8], src + u * 8);
            // V tile: 128 rows x 32 keys; 1 row per thread
            const bf16* vsrc = Vt + vbase + (size_t)tid * T_ + kbase;
#pragma unroll
            for (int u = 0; u < 4; ++u)
                lds_copy16(&Vs[tid][u * 8], vsrc + u * 8);
        }
        lds_copy_fence();
        __syncthreads();

        // S = Q' K'^T  (two 16-key tiles)
        v8f s[2];
#pragma unroll
        for (int j = 0; j < 2; ++j) {
            s[j] = v8f_zero();
#pragma unroll
            for (int kc = 0; kc < 6; ++kc) {
                v16bf bfr = *reinterpret_cast<const v16bf*>(
                    &Ks[j * 16 + ln][kc * 32 + sel * 16]);
                s[j] = __builtin_amdgcn_wmma_f32_16x16x32_bf16(
                    false, qf[kc], false, bfr, (short)0, s[j], false, false);
            }
        }

        // Online softmax (rows live in fixed 16-lane groups)
        float newm[8];
#pragma unroll
        for (int r = 0; r < 8; ++r) {
            int qrow = q0 + wave * 16 + r + 8 * sel;
            float mx = mrow[r];
#pragma unroll
            for (int j = 0; j < 2; ++j) {
                int key = kbase + j * 16 + ln;
                float sv = s[j][r] * scale;
                if (key > qrow) sv = -INFINITY;
                s[j][r] = sv;
                mx = fmaxf(mx, sv);
            }
#pragma unroll
            for (int msk = 1; msk < 16; msk <<= 1)
                mx = fmaxf(mx, __shfl_xor(mx, msk, 32));
            newm[r] = mx;
        }
#pragma unroll
        for (int r = 0; r < 8; ++r) {
            float alpha = __expf(mrow[r] - newm[r]);
            float rs = 0.f;
#pragma unroll
            for (int j = 0; j < 2; ++j) {
                float p = __expf(s[j][r] - newm[r]);
                s[j][r] = p;
                rs += p;
            }
#pragma unroll
            for (int msk = 1; msk < 16; msk <<= 1)
                rs += __shfl_xor(rs, msk, 32);
            lrow[r] = lrow[r] * alpha + rs;
            mrow[r] = newm[r];
#pragma unroll
            for (int i = 0; i < 8; ++i) o[i][r] *= alpha;
        }

        // Transpose P through per-wave LDS into A-fragment layout
#pragma unroll
        for (int r = 0; r < 8; ++r)
#pragma unroll
            for (int j = 0; j < 2; ++j)
                Ps[wave][r + 8 * sel][j * 16 + ln] = (bf16)s[j][r];
        v16bf pf = load_a_frag(&Ps[wave][ln][0], sel);

        // O += P (16x32) * V (32x16 per hd tile)
#pragma unroll
        for (int i = 0; i < 8; ++i) {
            v16bf vf = *reinterpret_cast<const v16bf*>(&Vs[i * 16 + ln][sel * 16]);
            o[i] = __builtin_amdgcn_wmma_f32_16x16x32_bf16(
                false, pf, false, vf, (short)0, o[i], false, false);
        }
        __syncthreads();
    }

    // Epilogue: normalize + store to (B,T,H*HD)
#pragma unroll
    for (int r = 0; r < 8; ++r) {
        float inv = 1.0f / lrow[r];
        int row = q0 + wave * 16 + r + 8 * sel;
        bf16* orow = Ao + ((size_t)(b * T_ + row)) * D_ + h * HD_;
#pragma unroll
        for (int i = 0; i < 8; ++i)
            orow[i * 16 + ln] = (bf16)(o[i][r] * inv);
    }
}

// ---------------------------------------------------------------------------
// Host launch
// ---------------------------------------------------------------------------
extern "C" void kernel_launch(void* const* d_in, const int* in_sizes, int n_in,
                              void* d_out, int out_size, void* d_ws, size_t ws_size,
                              hipStream_t stream) {
    const float* x    = (const float*)d_in[0];
    const float* Wqd  = (const float*)d_in[1];
    const float* Wqu  = (const float*)d_in[2];
    const float* Wqr  = (const float*)d_in[3];
    const float* Wkvd = (const float*)d_in[4];
    const float* Wku  = (const float*)d_in[5];
    const float* Wvu  = (const float*)d_in[6];
    const float* Wkr  = (const float*)d_in[7];
    const float* Wo   = (const float*)d_in[8];

    char* ws = (char*)d_ws;
    size_t off = 0;
    auto carve = [&](size_t elems) -> bf16* {
        bf16* p = (bf16*)(ws + off);
        off = (off + elems * sizeof(bf16) + 255) & ~(size_t)255;
        return p;
    };

    const size_t BT = (size_t)B_ * T_;              // 4096
    bf16* xb     = carve(BT * D_);
    bf16* Wqd_t  = carve((size_t)LAT_ * D_);
    bf16* Wqu_t  = carve((size_t)D_ * LAT_);
    bf16* Wqr_t  = carve((size_t)H_ * RD_ * LAT_);
    bf16* Wkvd_t = carve((size_t)LAT_ * D_);
    bf16* Wku_t  = carve((size_t)D_ * LAT_);
    bf16* Wvu_t  = carve((size_t)D_ * LAT_);
    bf16* Wkr_t  = carve((size_t)RD_ * D_);
    bf16* Wo_t   = carve((size_t)D_ * D_);
    bf16* ql     = carve(BT * LAT_);
    bf16* ckv    = carve(BT * LAT_);
    bf16* qtmp   = carve(BT * D_);
    bf16* qrtmp  = carve(BT * H_ * RD_);
    bf16* ktmp   = carve(BT * D_);
    bf16* vtmp   = carve(BT * D_);
    bf16* krtmp  = carve(BT * RD_);
    bf16* Qp     = carve((size_t)B_ * H_ * T_ * QKD_);
    bf16* Kp     = carve((size_t)B_ * H_ * T_ * QKD_);
    bf16* Vt     = carve((size_t)B_ * H_ * HD_ * T_);
    bf16* Ao     = carve(BT * D_);

    auto cvt = [&](const float* in, bf16* out, size_t n) {
        f32_to_bf16_kernel<<<dim3((unsigned)((n + 255) / 256)), dim3(256), 0, stream>>>(in, out, n);
    };
    auto tpose = [&](const float* in, bf16* out, int K, int N) {
        size_t n = (size_t)K * N;
        transpose_f32_to_bf16_kernel<<<dim3((unsigned)((n + 255) / 256)), dim3(256), 0, stream>>>(in, out, K, N);
    };

    cvt(x, xb, BT * D_);
    tpose(Wqd,  Wqd_t,  D_,   LAT_);
    tpose(Wqu,  Wqu_t,  LAT_, D_);
    tpose(Wqr,  Wqr_t,  LAT_, H_ * RD_);
    tpose(Wkvd, Wkvd_t, D_,   LAT_);
    tpose(Wku,  Wku_t,  LAT_, D_);
    tpose(Wvu,  Wvu_t,  LAT_, D_);
    tpose(Wkr,  Wkr_t,  D_,   RD_);
    tpose(Wo,   Wo_t,   D_,   D_);

    auto gemm_bf = [&](const bf16* A, const bf16* Bt, bf16* C, int M, int N, int K) {
        gemm_bf16_kernel<bf16><<<dim3(N / 64, M / 128), dim3(256), 0, stream>>>(A, Bt, C, M, N, K);
    };

    gemm_bf(xb,  Wqd_t,  ql,    (int)BT, LAT_,      D_);
    gemm_bf(xb,  Wkvd_t, ckv,   (int)BT, LAT_,      D_);
    gemm_bf(ql,  Wqu_t,  qtmp,  (int)BT, D_,        LAT_);
    gemm_bf(ql,  Wqr_t,  qrtmp, (int)BT, H_ * RD_,  LAT_);
    gemm_bf(ckv, Wku_t,  ktmp,  (int)BT, D_,        LAT_);
    gemm_bf(ckv, Wvu_t,  vtmp,  (int)BT, D_,        LAT_);
    gemm_bf(xb,  Wkr_t,  krtmp, (int)BT, RD_,       D_);

    pack_rope_kernel<<<dim3(T_, B_), dim3(256), 0, stream>>>(
        qtmp, qrtmp, ktmp, vtmp, krtmp, Qp, Kp, Vt);

    flash_attn_kernel<<<dim3(T_ / 64, H_, B_), dim3(128), 0, stream>>>(Qp, Kp, Vt, Ao);

    gemm_bf16_kernel<float><<<dim3(D_ / 64, (int)BT / 128), dim3(256), 0, stream>>>(
        Ao, Wo_t, (float*)d_out, (int)BT, D_, D_);
}
